// cluster_pooling_mil_53661321396715
// MI455X (gfx1250) — compile-verified, matrix operand order checked by reading, not run on hardware
//
#include <hip/hip_runtime.h>

// ---------------------------------------------------------------------------
// MI455X (gfx1250) cluster-pooling MIL head.
// Encoder GEMM (68.7 GFLOP) on v_wmma_f32_16x16x32_bf16. bags read (512 MB @
// 23.3 TB/s ~ 22 us) roughly balances bf16 WMMA compute. A operands loaded
// directly per-lane (each 32B group = exactly one use of a 128B line),
// distance-1 register prefetch; B operands pre-packed and DMA'd via
// global_load_async_to_lds_b128 (ASYNCcnt) into ping-pong LDS buffers.
// K-means / BN / head are small VALU kernels on L2-resident data (134 MB emb).
// ---------------------------------------------------------------------------

typedef __attribute__((ext_vector_type(16))) __bf16 v16bf;
typedef __attribute__((ext_vector_type(8)))  float  v8f;

#define ZDIM   256
#define FDIM   1024
#define NBAGS  32
#define NPTS   4096
#define NCLUST 4
#define MTILE  128

__device__ __forceinline__ unsigned short f2bf(float f) {
    unsigned u = __float_as_uint(f);
    unsigned r = u + 0x7FFFu + ((u >> 16) & 1u);   // round-to-nearest-even
    return (unsigned short)(r >> 16);
}

__device__ __forceinline__ v16bf mk16(uint4 a, uint4 b) {
    union { uint4 u[2]; v16bf v; } un;
    un.u[0] = a; un.u[1] = b;
    return un.v;
}

// pack 16 f32 (A-fragment slots 0..15) into v16bf
__device__ __forceinline__ v16bf cvt16(float4 a, float4 b, float4 c, float4 d) {
    union { unsigned short u[16]; v16bf v; } un;
    un.u[0]  = f2bf(a.x); un.u[1]  = f2bf(a.y); un.u[2]  = f2bf(a.z); un.u[3]  = f2bf(a.w);
    un.u[4]  = f2bf(b.x); un.u[5]  = f2bf(b.y); un.u[6]  = f2bf(b.z); un.u[7]  = f2bf(b.w);
    un.u[8]  = f2bf(c.x); un.u[9]  = f2bf(c.y); un.u[10] = f2bf(c.z); un.u[11] = f2bf(c.w);
    un.u[12] = f2bf(d.x); un.u[13] = f2bf(d.y); un.u[14] = f2bf(d.z); un.u[15] = f2bf(d.w);
    return un.v;
}

// --- pack W_enc (f32 [K=1024][N=256]) into per-lane WMMA B-fragment order ---
// Bp[((kt*16+nt)*32 + lane)*16 + s] = bf16( W[kt*32 + klocal(lane,s)][nt*16 + (lane&15)] )
__global__ __launch_bounds__(256) void prep_w(const float* __restrict__ W,
                                              unsigned short* __restrict__ Bp) {
    int idx  = blockIdx.x * 256 + threadIdx.x;      // 0..262143
    int s    = idx & 15;
    int lane = (idx >> 4) & 31;
    int frag = idx >> 9;                            // 0..511
    int nt   = frag & 15;
    int kt   = frag >> 4;
    int kl   = (lane < 16) ? ((s < 8) ? s : s + 8)
                           : ((s < 8) ? s + 8 : s + 16);
    int k = kt * 32 + kl;
    int n = nt * 16 + (lane & 15);
    Bp[idx] = f2bf(W[k * ZDIM + n]);
}

// issue one k-step's 16 KB of pre-packed B into LDS via CDNA5 async DMA
__device__ __forceinline__ void issueB(const unsigned short* __restrict__ Bp,
                                       int kt, unsigned short* dstBase, int t) {
    const unsigned short* src = Bp + (size_t)kt * 8192 + t * 32;   // 64 B / thread
    unsigned long long ga  = (unsigned long long)(uintptr_t)src;
    unsigned           lds = (unsigned)(uintptr_t)(dstBase + t * 32);
    #pragma unroll
    for (int i = 0; i < 4; ++i)
        asm volatile("global_load_async_to_lds_b128 %0, %1, off"
                     :: "v"(lds + i * 16), "v"(ga + (unsigned long long)(i * 16))
                     : "memory");
}

// --- encoder: emb = relu(bags @ W_enc + b_enc), WMMA bf16 ---
__global__ __launch_bounds__(256) void enc_gemm(const float* __restrict__ bags,
                                                const unsigned short* __restrict__ Bp,
                                                const float* __restrict__ b_enc,
                                                float* __restrict__ emb) {
    __shared__ __align__(16) unsigned short sB[2][16 * 512];   // ping-pong, 32 KB

    const int t    = threadIdx.x;
    const int w    = t >> 5;
    const int lane = t & 31;
    const int hi   = lane >> 4;
    const long row0 = (long)blockIdx.x * MTILE;

    v8f acc[16] = {};

    // per-lane A row base: lane owns row (w*16 + lane&15), k-groups hi*8 / hi*8+16
    const float* arow = bags + (row0 + w * 16 + (lane & 15)) * FDIM + hi * 8;
    const int boff = lane * 16;   // ushort offset of this lane's B-frag piece

    // preload kt=0: B via async DMA, A via direct b128 loads into registers
    issueB(Bp, 0, &sB[0][0], t);
    float4 a0 = *(const float4*)(arow + 0);
    float4 a1 = *(const float4*)(arow + 4);
    float4 a2 = *(const float4*)(arow + 16);
    float4 a3 = *(const float4*)(arow + 20);

    for (int kt = 0; kt < 32; ++kt) {
        const int cur = kt & 1;
        // own async B writes for buffer `cur` complete, then block-wide sync
        asm volatile("s_wait_asynccnt 0x0" ::: "memory");
        __syncthreads();

        // prefetch next k-step (B -> other LDS buffer, A -> registers)
        float4 n0 = a0, n1 = a1, n2 = a2, n3 = a3;
        if (kt + 1 < 32) {
            issueB(Bp, kt + 1, &sB[cur ^ 1][0], t);
            const float* an = arow + (kt + 1) * 32;
            n0 = *(const float4*)(an + 0);
            n1 = *(const float4*)(an + 4);
            n2 = *(const float4*)(an + 16);
            n3 = *(const float4*)(an + 20);
        }

        v16bf av = cvt16(a0, a1, a2, a3);   // VALU, co-executes with XDL WMMA

        // software-pipelined B-fragment feed (alternating register sets)
        uint4 p0 = *(const uint4*)&sB[cur][boff];
        uint4 p1 = *(const uint4*)&sB[cur][boff + 8];
        #pragma unroll
        for (int nt = 0; nt < 16; ++nt) {
            uint4 q0 = p0, q1 = p1;
            if (nt + 1 < 16) {
                q0 = *(const uint4*)&sB[cur][(nt + 1) * 512 + boff];
                q1 = *(const uint4*)&sB[cur][(nt + 1) * 512 + boff + 8];
            }
            acc[nt] = __builtin_amdgcn_wmma_f32_16x16x32_bf16(
                false, av, false, mk16(p0, p1), (short)0, acc[nt], false, false);
            p0 = q0; p1 = q1;
        }

        a0 = n0; a1 = n1; a2 = n2; a3 = n3;
    }

    // epilogue: bias + relu; C layout: VGPR r -> M = r (lanes 0-15) / r+8 (16-31)
    const int rbase = w * 16 + hi * 8;
    #pragma unroll
    for (int nt = 0; nt < 16; ++nt) {
        int col = nt * 16 + (lane & 15);
        float bias = b_enc[col];
        #pragma unroll
        for (int r = 0; r < 8; ++r) {
            float v = acc[nt][r] + bias;
            v = v > 0.f ? v : 0.f;
            emb[(row0 + rbase + r) * ZDIM + col] = v;
        }
    }
}

// --- kmeans: initial centroids = emb rows {0,1024,2048,3072} of each bag ---
__global__ __launch_bounds__(1024) void km_init(const float* __restrict__ emb,
                                                float* __restrict__ cent0) {
    int t = threadIdx.x, bag = blockIdx.x;
    int c = t >> 8, z = t & 255;
    cent0[bag * 1024 + t] = emb[((size_t)bag * NPTS + c * 1024) * ZDIM + z];
}

// --- kmeans assign + partial accumulation (grid: 16 segs x 32 bags) ---
__global__ __launch_bounds__(256) void km_assign(const float* __restrict__ emb,
                                                 const float* __restrict__ cent,
                                                 float* __restrict__ sums,
                                                 float* __restrict__ cnts) {
    __shared__ float         sc[NCLUST * ZDIM];
    __shared__ unsigned char slab[256];
    const int t   = threadIdx.x;
    const int seg = blockIdx.x;
    const int bag = blockIdx.y;

    #pragma unroll
    for (int i = 0; i < NCLUST; ++i)
        sc[i * ZDIM + t] = cent[bag * 1024 + i * ZDIM + t];
    __syncthreads();

    const float* e = emb + ((size_t)bag * NPTS + seg * 256 + t) * ZDIM;
    float d0 = 0.f, d1 = 0.f, d2 = 0.f, d3 = 0.f;
    for (int z = 0; z < ZDIM; z += 4) {
        float4 ev = *(const float4*)(e + z);
        #pragma unroll
        for (int j = 0; j < 4; ++j) {
            float x = (&ev.x)[j];
            float u0 = x - sc[z + j];
            float u1 = x - sc[256 + z + j];
            float u2 = x - sc[512 + z + j];
            float u3 = x - sc[768 + z + j];
            d0 += u0 * u0; d1 += u1 * u1; d2 += u2 * u2; d3 += u3 * u3;
        }
    }
    int lab = 0; float dm = d0;
    if (d1 < dm) { dm = d1; lab = 1; }
    if (d2 < dm) { dm = d2; lab = 2; }
    if (d3 < dm) { dm = d3; lab = 3; }
    slab[t] = (unsigned char)lab;
    __syncthreads();

    // transpose accumulation: thread t owns dimension z=t; coalesced emb reads
    const float* ebase = emb + ((size_t)bag * NPTS + seg * 256) * ZDIM;
    float s0 = 0.f, s1 = 0.f, s2 = 0.f, s3 = 0.f;
    for (int p = 0; p < 256; ++p) {
        float v = ebase[(size_t)p * ZDIM + t];
        int l = slab[p];
        s0 += (l == 0) ? v : 0.f;
        s1 += (l == 1) ? v : 0.f;
        s2 += (l == 2) ? v : 0.f;
        s3 += (l == 3) ? v : 0.f;
    }
    atomicAdd(&sums[bag * 1024 + 0 * 256 + t], s0);
    atomicAdd(&sums[bag * 1024 + 1 * 256 + t], s1);
    atomicAdd(&sums[bag * 1024 + 2 * 256 + t], s2);
    atomicAdd(&sums[bag * 1024 + 3 * 256 + t], s3);
    if (t < NCLUST) {
        int cnt = 0;
        for (int p = 0; p < 256; ++p) cnt += (slab[p] == t);
        atomicAdd(&cnts[bag * NCLUST + t], (float)cnt);
    }
}

// --- centroid update: cnt>0 ? sum/max(cnt,1) : old (exact reference rule) ---
__global__ __launch_bounds__(1024) void km_update(const float* __restrict__ sums,
                                                  const float* __restrict__ cnts,
                                                  const float* __restrict__ centIn,
                                                  float* __restrict__ centOut) {
    int t = threadIdx.x, bag = blockIdx.x;
    float cn = cnts[bag * NCLUST + (t >> 8)];
    float s  = sums[bag * 1024 + t];
    float ci = centIn[bag * 1024 + t];
    centOut[bag * 1024 + t] = (cn > 0.f) ? s / fmaxf(cn, 1.f) : ci;
}

// --- pooled means: cnt>0 ? sum/max(cnt,1) : 0 ---
__global__ __launch_bounds__(1024) void km_pool(const float* __restrict__ sums,
                                                const float* __restrict__ cnts,
                                                float* __restrict__ pooled) {
    int t = threadIdx.x, bag = blockIdx.x;
    float cn = cnts[bag * NCLUST + (t >> 8)];
    float s  = sums[bag * 1024 + t];
    pooled[bag * 1024 + t] = (cn > 0.f) ? s / fmaxf(cn, 1.f) : 0.f;
}

// --- BatchNorm over batch dim (training stats), gamma/beta ---
__global__ __launch_bounds__(256) void bn_kernel(const float* __restrict__ pooled,
                                                 const float* __restrict__ gamma,
                                                 const float* __restrict__ beta,
                                                 float* __restrict__ h) {
    int f = blockIdx.x * 256 + threadIdx.x;   // 0..1023
    float m = 0.f;
    for (int b = 0; b < NBAGS; ++b) m += pooled[b * 1024 + f];
    m *= (1.f / NBAGS);
    float v = 0.f;
    for (int b = 0; b < NBAGS; ++b) {
        float d = pooled[b * 1024 + f] - m;
        v += d * d;
    }
    v *= (1.f / NBAGS);
    float inv = rsqrtf(v + 1e-5f);
    float g = gamma[f], be = beta[f];
    for (int b = 0; b < NBAGS; ++b)
        h[b * 1024 + f] = (pooled[b * 1024 + f] - m) * inv * g + be;
}

// --- head: out[32][2] = h @ W_head + b_head ---
__global__ __launch_bounds__(64) void head_kernel(const float* __restrict__ h,
                                                  const float* __restrict__ Wh,
                                                  const float* __restrict__ bh,
                                                  float* __restrict__ out) {
    int t = threadIdx.x;
    if (t >= 64) return;
    int b = t >> 1, o = t & 1;
    float a = bh[o];
    for (int f = 0; f < 1024; ++f)
        a += h[b * 1024 + f] * Wh[f * 2 + o];
    out[b * 2 + o] = a;
}

extern "C" void kernel_launch(void* const* d_in, const int* in_sizes, int n_in,
                              void* d_out, int out_size, void* d_ws, size_t ws_size,
                              hipStream_t stream) {
    (void)in_sizes; (void)n_in; (void)out_size; (void)ws_size;
    const float* bags   = (const float*)d_in[0];
    const float* W_enc  = (const float*)d_in[1];
    const float* b_enc  = (const float*)d_in[2];
    const float* gamma  = (const float*)d_in[3];
    const float* beta   = (const float*)d_in[4];
    const float* W_head = (const float*)d_in[5];
    const float* b_head = (const float*)d_in[6];
    float* out = (float*)d_out;

    char* ws = (char*)d_ws;
    size_t off = 0;
    float* emb = (float*)(ws + off);              off += (size_t)NBAGS * NPTS * ZDIM * 4;  // 134 MB
    unsigned short* Bp = (unsigned short*)(ws + off); off += (size_t)FDIM * ZDIM * 2;       // 512 KB
    float* cent = (float*)(ws + off);             off += (size_t)11 * NBAGS * 1024 * 4;     // cent[0..10]
    float* sums = (float*)(ws + off);             off += (size_t)11 * NBAGS * 1024 * 4;     // sums[0..10]
    float* cnts = (float*)(ws + off);             off += (size_t)11 * NBAGS * NCLUST * 4;
    float* pooled = (float*)(ws + off);           off += (size_t)NBAGS * 1024 * 4;
    float* h = (float*)(ws + off);                off += (size_t)NBAGS * 1024 * 4;

    // zero sums + cnts (contiguous region), graph-capturable
    hipMemsetAsync(sums, 0,
                   (size_t)11 * NBAGS * 1024 * 4 + (size_t)11 * NBAGS * NCLUST * 4,
                   stream);

    prep_w  <<<1024, 256, 0, stream>>>(W_enc, Bp);
    enc_gemm<<<1024, 256, 0, stream>>>(bags, Bp, b_enc, emb);
    km_init <<<NBAGS, 1024, 0, stream>>>(emb, cent);

    for (int i = 0; i < 10; ++i) {
        km_assign<<<dim3(16, NBAGS), 256, 0, stream>>>(
            emb, cent + (size_t)i * NBAGS * 1024,
            sums + (size_t)i * NBAGS * 1024, cnts + (size_t)i * NBAGS * NCLUST);
        km_update<<<NBAGS, 1024, 0, stream>>>(
            sums + (size_t)i * NBAGS * 1024, cnts + (size_t)i * NBAGS * NCLUST,
            cent + (size_t)i * NBAGS * 1024, cent + (size_t)(i + 1) * NBAGS * 1024);
    }
    // final assignment with cent[10] doubles as pooling accumulation (slot 10)
    km_assign<<<dim3(16, NBAGS), 256, 0, stream>>>(
        emb, cent + (size_t)10 * NBAGS * 1024,
        sums + (size_t)10 * NBAGS * 1024, cnts + (size_t)10 * NBAGS * NCLUST);
    km_pool<<<NBAGS, 1024, 0, stream>>>(
        sums + (size_t)10 * NBAGS * 1024, cnts + (size_t)10 * NBAGS * NCLUST, pooled);

    bn_kernel  <<<4, 256, 0, stream>>>(pooled, gamma, beta, h);
    head_kernel<<<1, 64, 0, stream>>>(h, W_head, b_head, out);
}
